// SchNet_87230785782288
// MI455X (gfx1250) — matrix operand order repaired and checked by reading, compiled, not run on hardware
//
#include <hip/hip_runtime.h>

typedef __attribute__((ext_vector_type(16))) __bf16 v16bf;
typedef __attribute__((ext_vector_type(8)))  float  v8f;

#define NN 80000
#define EE 1280000
#define BB 256
#define HH 128
#define RR 64
#define KK 5
#define CUT 5.0f

// fast silu: x * rcp(1+exp(-x)) -> v_exp_f32 + v_add + v_rcp_f32 + v_mul (no IEEE div sequence)
__device__ __forceinline__ float silu_f(float x) {
  return x * __builtin_amdgcn_rcpf(1.0f + __expf(-x));
}

__device__ __forceinline__ v8f wmma_bf16(v16bf a, v16bf b, v8f c) {
  return __builtin_amdgcn_wmma_f32_16x16x32_bf16(false, a, false, b, (short)0, c, false, false);
}

// B fragment: weights stored bf16 column-major (col*ldk + k), packed K pairs contiguous.
// ISA dense B 32x16 layout: lanes 0-15 hold K=kbase+0..7,16..23 (VGPR j -> K=(j<4?0:16)+(j&3)*2),
// lanes 16-31 hold K=kbase+8..15,24..31. n = lane&15.
__device__ __forceinline__ v16bf load_B_frag(const __bf16* __restrict__ w, int ldk,
                                             int lane, int kbase, int nbase) {
  int n = lane & 15, g = (lane >> 4) & 1;
  const __bf16* p = w + (size_t)(nbase + n) * ldk + kbase + g * 8;
  v16bf b;
#pragma unroll
  for (int j = 0; j < 8; ++j) { int o = (j < 4) ? 0 : 8; b[2 * j] = p[2 * j + o]; b[2 * j + 1] = p[2 * j + 1 + o]; }
  return b;
}

// A 16x32 bf16 layout: m = lane&15; VGPR j holds K = (j<4?0:16) + (lane>=16?8:0) + (j&3)*2
#define A_KK(kc, j, g) ((kc) * 32 + (((j) & 4) ? 16 : 0) + (g) * 8 + ((j) & 3) * 2)

// ---------------- weight conversion to bf16 column-major ----------------
__global__ void prep_weights(const float* __restrict__ w1, const float* __restrict__ w2,
                             const float* __restrict__ wd, const float* __restrict__ wb,
                             __bf16* __restrict__ w1b, __bf16* __restrict__ w2b,
                             __bf16* __restrict__ wdb, __bf16* __restrict__ wbb) {
  int i = blockIdx.x * blockDim.x + threadIdx.x;
  if (i < KK * RR * HH) {
    int k = i / (RR * HH), rem = i % (RR * HH), r = rem / HH, h = rem % HH;
    w1b[k * RR * HH + h * RR + r] = (__bf16)w1[i];
  }
  if (i < KK * HH * HH) {
    int k = i / (HH * HH), rem = i % (HH * HH), r = rem / HH, h = rem % HH;
    int o = k * HH * HH + h * HH + r;
    w2b[o] = (__bf16)w2[i];
    wdb[o] = (__bf16)wd[i];
    wbb[o] = (__bf16)wb[i];
  }
}

__global__ void init_x(const float* __restrict__ emb, const int* __restrict__ z,
                       float* __restrict__ x) {
  int i = blockIdx.x * blockDim.x + threadIdx.x;
  if (i < NN * HH) { int n = i / HH, h = i % HH; x[i] = emb[z[n] * HH + h]; }
}

__global__ void zero_f32(float* __restrict__ p, int n) {
  int i = blockIdx.x * blockDim.x + threadIdx.x;
  if (i < n) p[i] = 0.0f;
}

// ---------------- node-level GEMM: out = [silu?](A) @ W (+bias) (+=out) ----------------
template <int SILU, int ACC>
__global__ __launch_bounds__(128) void node_gemm(const float* __restrict__ A,
                                                 const __bf16* __restrict__ W,
                                                 const float* __restrict__ bias,
                                                 float* __restrict__ out) {
  int lane = threadIdx.x & 31, wv = threadIdx.x >> 5;
  int n0 = (blockIdx.x * 4 + wv) * 16;
  int m = lane & 15, g = lane >> 4;
  v8f acc[8] = {};
  const float* rowp = A + (size_t)(n0 + m) * HH;
#pragma unroll
  for (int kc = 0; kc < 4; ++kc) {
    v16bf a;
#pragma unroll
    for (int j = 0; j < 8; ++j) {
      int kk = A_KK(kc, j, g);
      float f0 = rowp[kk], f1 = rowp[kk + 1];
      if (SILU) { f0 = silu_f(f0); f1 = silu_f(f1); }
      a[2 * j] = (__bf16)f0; a[2 * j + 1] = (__bf16)f1;
    }
#pragma unroll
    for (int nt = 0; nt < 8; ++nt) {
      v16bf b = load_B_frag(W, HH, lane, kc * 32, nt * 16);
      acc[nt] = wmma_bf16(a, b, acc[nt]);
    }
  }
#pragma unroll
  for (int nt = 0; nt < 8; ++nt) {
    int col = nt * 16 + m;
    float bv = (bias != nullptr) ? bias[col] : 0.0f;
#pragma unroll
    for (int j = 0; j < 8; ++j) {          // D layout: VGPR j -> row j (+8 for lanes 16-31)
      size_t idx = (size_t)(n0 + j + g * 8) * HH + col;
      float r = acc[nt][j] + bv;
      if (ACC) out[idx] += r; else out[idx] = r;
    }
  }
}

// ---------------- fused edge kernel: rbf -> MLP(WMMA) -> cutoff*gather -> scatter ----------------
__global__ __launch_bounds__(128) void edge_msg(const float* __restrict__ pos,
                                                const int* __restrict__ src,
                                                const int* __restrict__ dst,
                                                const float* __restrict__ gamma,
                                                const float* __restrict__ b1,
                                                const float* __restrict__ b2,
                                                const __bf16* __restrict__ w1b,
                                                const __bf16* __restrict__ w2b,
                                                const float* __restrict__ xd,
                                                float* __restrict__ v, int k) {
  __shared__ float s_hid[4][16 * 136];   // padded to dodge bank conflicts
  __shared__ float s_d[4][16], s_C[4][16];
  __shared__ int   s_s[4][16], s_t[4][16];
  int lane = threadIdx.x & 31, wv = threadIdx.x >> 5;
  int e0 = (blockIdx.x * 4 + wv) * 16;
  int m = lane & 15, g = lane >> 4;

  if (lane < 16) {
    int e = e0 + lane;
    int s = src[e], t = dst[e];
    float dx = pos[3 * s + 0] - pos[3 * t + 0];
    float dy = pos[3 * s + 1] - pos[3 * t + 1];
    float dz = pos[3 * s + 2] - pos[3 * t + 2];
    float d = sqrtf(dx * dx + dy * dy + dz * dz);
    float xc = d * (1.0f / CUT);
    float C = (xc < 1.0f) ? 0.5f * (__cosf(3.14159265358979f * xc) + 1.0f) : 0.0f;
    s_d[wv][lane] = d; s_C[wv][lane] = C; s_s[wv][lane] = s; s_t[wv][lane] = t;
  }
  __syncthreads();

  float dm = s_d[wv][m];
  float gam = gamma[k];
  const float cstep = CUT / (RR - 1);
  const __bf16* W1 = w1b + (size_t)k * RR * HH;
  v8f acc[8] = {};
#pragma unroll
  for (int kc = 0; kc < 2; ++kc) {         // K = 64 over two 32-chunks
    v16bf a;                               // rbf generated directly in A-frag layout
#pragma unroll
    for (int j = 0; j < 8; ++j) {
      int kk = A_KK(kc, j, g);
      float t0 = dm - cstep * kk;
      float t1 = dm - cstep * (kk + 1);
      a[2 * j]     = (__bf16)__expf(-gam * t0 * t0);
      a[2 * j + 1] = (__bf16)__expf(-gam * t1 * t1);
    }
#pragma unroll
    for (int nt = 0; nt < 8; ++nt) {
      v16bf b = load_B_frag(W1, RR, lane, kc * 32, nt * 16);
      acc[nt] = wmma_bf16(a, b, acc[nt]);
    }
  }

  const float* B1 = b1 + k * HH;           // hidden = silu(gemm1 + b1) -> LDS (transpose hop)
#pragma unroll
  for (int nt = 0; nt < 8; ++nt) {
    float bv = B1[nt * 16 + m];
#pragma unroll
    for (int j = 0; j < 8; ++j) {
      int row = j + g * 8;
      s_hid[wv][row * 136 + nt * 16 + m] = silu_f(acc[nt][j] + bv);
    }
  }
  __syncthreads();

  const __bf16* W2 = w2b + (size_t)k * HH * HH;
  const float* hrow = &s_hid[wv][m * 136];
  v8f acc2[8] = {};
#pragma unroll
  for (int kc = 0; kc < 4; ++kc) {         // K = 128
    v16bf a;
#pragma unroll
    for (int j = 0; j < 8; ++j) {
      int kk = A_KK(kc, j, g);
      a[2 * j] = (__bf16)hrow[kk]; a[2 * j + 1] = (__bf16)hrow[kk + 1];
    }
#pragma unroll
    for (int nt = 0; nt < 8; ++nt) {
      v16bf b = load_B_frag(W2, HH, lane, kc * 32, nt * 16);
      acc2[nt] = wmma_bf16(a, b, acc2[nt]);
    }
  }

  const float* B2 = b2 + k * HH;           // W = (gemm2+b2)*C; msg = W * xd[src]; v[dst] += msg
#pragma unroll
  for (int nt = 0; nt < 8; ++nt) {
    int col = nt * 16 + m;
    float bv = B2[col];
#pragma unroll
    for (int j = 0; j < 8; ++j) {
      int row = j + g * 8;
      float Wv = (acc2[nt][j] + bv) * s_C[wv][row];
      int s = s_s[wv][row], t = s_t[wv][row];
      float msg = Wv * xd[(size_t)s * HH + col];
      atomicAdd(&v[(size_t)t * HH + col], msg);
    }
  }
}

// ---------------- pooling (batch sorted -> run-length, few atomics) ----------------
__global__ __launch_bounds__(128) void pool_kernel(const float* __restrict__ x,
                                                   const int* __restrict__ batch,
                                                   float* __restrict__ pooled,
                                                   float* __restrict__ counts) {
  const int CH = 313;
  int n0 = blockIdx.x * CH;
  int n1 = min(NN, n0 + CH);
  if (n0 >= n1) return;
  int h = threadIdx.x;
  float acc = 0.0f; int cur = batch[n0];
  for (int n = n0; n < n1; ++n) {
    int b = batch[n];
    if (b != cur) { atomicAdd(&pooled[cur * HH + h], acc); acc = 0.0f; cur = b; }
    acc += x[(size_t)n * HH + h];
  }
  atomicAdd(&pooled[cur * HH + h], acc);
  if (h == 0) {
    float cnt = 0.0f; cur = batch[n0];
    for (int n = n0; n < n1; ++n) {
      int b = batch[n];
      if (b != cur) { atomicAdd(&counts[cur], cnt); cnt = 0.0f; cur = b; }
      cnt += 1.0f;
    }
    atomicAdd(&counts[cur], cnt);
  }
}

// ---------------- FiLM + readout, one block per graph ----------------
__global__ __launch_bounds__(128) void readout_kernel(const float* __restrict__ pooled,
    const float* __restrict__ counts, const float* __restrict__ T,
    const float* __restrict__ fw1, const float* __restrict__ fb1,
    const float* __restrict__ fw2, const float* __restrict__ fb2,
    const float* __restrict__ rw1, const float* __restrict__ rb1,
    const float* __restrict__ rw2, const float* __restrict__ rb2,
    float* __restrict__ y) {
  __shared__ float st1[HH], sh[HH], red[HH];
  int b = blockIdx.x, h = threadIdx.x;
  float t = T[b];
  st1[h] = silu_f(t * fw1[h] + fb1[h]);
  __syncthreads();
  float gsum = fb2[h], bsum = fb2[HH + h];
  for (int i = 0; i < HH; ++i) {
    float ti = st1[i];
    gsum += ti * fw2[i * 2 * HH + h];
    bsum += ti * fw2[i * 2 * HH + HH + h];
  }
  float pool = pooled[b * HH + h] * __builtin_amdgcn_rcpf(fmaxf(counts[b], 1.0f));
  sh[h] = gsum * pool + bsum;
  __syncthreads();
  float us = rb1[h];
  for (int i = 0; i < HH; ++i) us += sh[i] * rw1[i * HH + h];
  red[h] = silu_f(us) * rw2[h];
  __syncthreads();
  for (int s = 64; s > 0; s >>= 1) { if (h < s) red[h] += red[h + s]; __syncthreads(); }
  if (h == 0) y[b] = red[0] + rb2[0];
}

extern "C" void kernel_launch(void* const* d_in, const int* in_sizes, int n_in,
                              void* d_out, int out_size, void* d_ws, size_t ws_size,
                              hipStream_t stream) {
  const float* emb  = (const float*)d_in[0];
  const float* w1   = (const float*)d_in[1];
  const float* b1   = (const float*)d_in[2];
  const float* w2   = (const float*)d_in[3];
  const float* b2   = (const float*)d_in[4];
  const float* wd   = (const float*)d_in[5];
  const float* wb   = (const float*)d_in[6];
  const float* bbv  = (const float*)d_in[7];
  const float* gam  = (const float*)d_in[8];
  const float* fw1  = (const float*)d_in[9];
  const float* fb1  = (const float*)d_in[10];
  const float* fw2  = (const float*)d_in[11];
  const float* fb2  = (const float*)d_in[12];
  const float* rw1  = (const float*)d_in[13];
  const float* rb1  = (const float*)d_in[14];
  const float* rw2  = (const float*)d_in[15];
  const float* rb2  = (const float*)d_in[16];
  const float* pos  = (const float*)d_in[17];
  const float* T    = (const float*)d_in[18];
  const int*   z    = (const int*)d_in[19];
  const int*   bat  = (const int*)d_in[20];
  const int*   eidx = (const int*)d_in[21];
  const int* srcp = eidx;
  const int* dstp = eidx + EE;

  char* ws = (char*)d_ws;
  size_t off = 0;
  auto take = [&](size_t bytes) { size_t o = off; off += (bytes + 255) & ~(size_t)255; return o; };
  float*  x    = (float*)(ws + take(sizeof(float) * NN * HH));
  float*  xd   = (float*)(ws + take(sizeof(float) * NN * HH));
  float*  vbuf = (float*)(ws + take(sizeof(float) * NN * HH));
  __bf16* w1b  = (__bf16*)(ws + take(sizeof(__bf16) * KK * RR * HH));
  __bf16* w2b  = (__bf16*)(ws + take(sizeof(__bf16) * KK * HH * HH));
  __bf16* wdb  = (__bf16*)(ws + take(sizeof(__bf16) * KK * HH * HH));
  __bf16* wbb  = (__bf16*)(ws + take(sizeof(__bf16) * KK * HH * HH));
  float*  pooled = (float*)(ws + take(sizeof(float) * BB * HH));
  float*  counts = (float*)(ws + take(sizeof(float) * BB));

  prep_weights<<<(KK * HH * HH + 255) / 256, 256, 0, stream>>>(w1, w2, wd, wb, w1b, w2b, wdb, wbb);
  init_x<<<(NN * HH + 255) / 256, 256, 0, stream>>>(emb, z, x);

  for (int k = 0; k < KK; ++k) {
    node_gemm<0, 0><<<NN / 64, 128, 0, stream>>>(x, wdb + (size_t)k * HH * HH, nullptr, xd);
    zero_f32<<<(NN * HH + 255) / 256, 256, 0, stream>>>(vbuf, NN * HH);
    edge_msg<<<EE / 64, 128, 0, stream>>>(pos, srcp, dstp, gam, b1, b2, w1b, w2b, xd, vbuf, k);
    node_gemm<1, 1><<<NN / 64, 128, 0, stream>>>(vbuf, wbb + (size_t)k * HH * HH, bbv + k * HH, x);
  }

  zero_f32<<<(BB * HH + 255) / 256, 256, 0, stream>>>(pooled, BB * HH);
  zero_f32<<<(BB + 255) / 256, 256, 0, stream>>>(counts, BB);
  pool_kernel<<<256, 128, 0, stream>>>(x, bat, pooled, counts);
  readout_kernel<<<BB, 128, 0, stream>>>(pooled, counts, T, fw1, fb1, fw2, fb2,
                                         rw1, rb1, rw2, rb2, (float*)d_out);
}
